// DistancePoolChebNet_52613349376609
// MI455X (gfx1250) — compile-verified, matrix operand order checked by reading, not run on hardware
//
#include <hip/hip_runtime.h>
#include <hip/hip_bf16.h>
#include <stdint.h>

// ---------------------------------------------------------------------------
// DistancePoolChebNet on MI455X (gfx1250, wave32, WMMA + double-buffered
// async global->LDS copies). GEMMs -> v_wmma_f32_16x16x32_f16 fed by
// global_load_async_to_lds_b128; edge prop -> per-jet LDS + ds_add_f32.
// ---------------------------------------------------------------------------

constexpr int B_  = 1024;
constexpr int P_  = 150;
constexpr int N_  = B_ * P_;        // 153600
constexpr int DEG_ = 16;
constexpr int E_  = N_ * DEG_;      // 2457600
constexpr int H_  = 256;
constexpr int IN_ = 3;
constexpr int NC_ = 10;

typedef __attribute__((ext_vector_type(16))) _Float16 v16h;
typedef __attribute__((ext_vector_type(8)))  _Float16 v8h;
typedef __attribute__((ext_vector_type(8)))  float    v8f;

// issue one per-lane 16-byte async global->LDS copy (ASYNCcnt-tracked)
__device__ __forceinline__ void async_copy16(uint32_t lds_off, const void* gptr) {
    asm volatile("global_load_async_to_lds_b128 %0, %1, off"
                 :: "v"(lds_off), "v"((uint64_t)(uintptr_t)gptr) : "memory");
}

// ------------------------------- utility kernels ---------------------------

__global__ void __launch_bounds__(256) zero_kernel(float* __restrict__ p, int n) {
    int i = blockIdx.x * 256 + threadIdx.x;
    if (i < n) p[i] = 0.f;
}

__global__ void __launch_bounds__(256) deg_kernel(const int* __restrict__ src,
                                                  const int* __restrict__ tgt,
                                                  float* __restrict__ deg) {
    int e = blockIdx.x * 256 + threadIdx.x;
    if (e < E_) {
        if (src[e] != tgt[e]) atomicAdd(&deg[src[e]], 1.0f);
    }
}

__global__ void __launch_bounds__(256) lapw_kernel(const int* __restrict__ src,
                                                   const int* __restrict__ tgt,
                                                   const float* __restrict__ deg,
                                                   float* __restrict__ lw) {
    int e = blockIdx.x * 256 + threadIdx.x;
    if (e < E_) {
        int s = src[e], t = tgt[e];
        if (s == t) { lw[e] = 0.f; return; }
        float ds = deg[s], dt = deg[t];
        float is = ds > 0.f ? rsqrtf(ds) : 0.f;
        float it = dt > 0.f ? rsqrtf(dt) : 0.f;
        lw[e] = -(is * it);
    }
}

__global__ void __launch_bounds__(256) bias_init_kernel(float* __restrict__ out,
                                                        const float* __restrict__ bias) {
    size_t i = (size_t)blockIdx.x * 256 + threadIdx.x;   // grid = N*H/256
    out[i] = bias[i & (H_ - 1)];
}

// layer-1 small GEMM: out[n,c] += sum_{j<3} T[n,j] * Wk[j,c]
__global__ void __launch_bounds__(256) gemm3_kernel(const float* __restrict__ T,
                                                    const float* __restrict__ Wk,
                                                    float* __restrict__ out) {
    size_t i = (size_t)blockIdx.x * 256 + threadIdx.x;
    int c = (int)(i & (H_ - 1));
    size_t n = i >> 8;
    out[i] += T[n * IN_ + 0] * Wk[0 * H_ + c]
            + T[n * IN_ + 1] * Wk[1 * H_ + c]
            + T[n * IN_ + 2] * Wk[2 * H_ + c];
}

// convert one layer's weights [6][k=256][c=256] f32 -> transposed f16 [6][c][k]
__global__ void __launch_bounds__(256) wtrans_kernel(const float* __restrict__ W,
                                                     _Float16* __restrict__ Wt) {
    size_t i = (size_t)blockIdx.x * 256 + threadIdx.x;   // over 6*65536
    int s = (int)(i >> 16);
    int c = (int)((i >> 8) & 255);   // output row  (column of W)
    int k = (int)(i & 255);          // output col  (row of W)
    Wt[i] = (_Float16)W[(size_t)s * H_ * H_ + (size_t)k * H_ + c];
}

// ------------------- Chebyshev propagation (per-jet, LDS) ------------------
// out = alpha * L_hat @ in  [- tprev]   over a 32-channel chunk per block.
__global__ void __launch_bounds__(256)
prop_kernel(const float* __restrict__ in, const float* __restrict__ tprev,
            float* __restrict__ out, const int* __restrict__ src,
            const int* __restrict__ tgt, const float* __restrict__ lw,
            int dim, float alpha) {
    extern __shared__ float smem[];
    const int jet   = blockIdx.x;
    const int c0    = blockIdx.y * 32;
    const int CH    = (dim - c0) < 32 ? (dim - c0) : 32;
    float* hin  = smem;
    float* hout = smem + P_ * CH;
    const int nbase = jet * P_;
    const int tid   = threadIdx.x;
    const int tot   = P_ * CH;

    for (int idx = tid; idx < tot; idx += 256) {
        int node = idx / CH, ch = idx - node * CH;
        hin[idx]  = in[(size_t)(nbase + node) * dim + c0 + ch];
        hout[idx] = 0.f;
    }
    __syncthreads();

    const int ebase = jet * P_ * DEG_;
    for (int e = tid; e < P_ * DEG_; e += 256) {
        const int eid = ebase + e;
        const float w = lw[eid];
        const int sl = (src[eid] - nbase) * CH;
        const int tl = (tgt[eid] - nbase) * CH;
        for (int ch = 0; ch < CH; ++ch)
            atomicAdd(&hout[tl + ch], w * hin[sl + ch]);   // ds_add_f32
    }
    __syncthreads();

    for (int idx = tid; idx < tot; idx += 256) {
        int node = idx / CH, ch = idx - node * CH;
        float v = alpha * hout[idx];
        if (tprev) v -= tprev[(size_t)(nbase + node) * dim + c0 + ch];
        out[(size_t)(nbase + node) * dim + c0 + ch] = v;
    }
}

// --------------------------- WMMA GEMM (H=256) -----------------------------
// out[N,256] += T[N,256] @ W[256,256]; Wt is f16 transposed [c][k].
// Block: 256 thr (8 waves), tile 64 rows x 256 cols; wave: 16 x 128 (8 tiles).
// A (f32) and B (f16) K-slices staged with GLOBAL_LOAD_ASYNC_TO_LDS_B128,
// double-buffered: next slice's copies run behind the current slice's WMMAs.
// Each wave issues 6 async instrs per slice -> s_wait_asynccnt 6 separates
// current (done) from next (in flight), relying on in-order async completion.
__global__ void __launch_bounds__(256)
wmma_gemm256(const float* __restrict__ T, const _Float16* __restrict__ Wt,
             float* __restrict__ out) {
    __shared__ __attribute__((aligned(16))) float    Asub[2][64 * 32];   // 16 KB f32 [row][k]
    __shared__ __attribute__((aligned(16))) _Float16 Bsub[2][256 * 32];  // 32 KB f16 [col][k]

    const int tid     = threadIdx.x;
    const int lane    = tid & 31;
    const int wave    = tid >> 5;
    const int rowBase = blockIdx.x * 64;
    const int mBase   = (wave >> 1) * 16;           // 0,16,32,48
    const int nBase   = (wave & 1) * 128;           // 0 / 128
    const int kb0     = (lane < 16) ? 0 : 8;        // half-wave K split (ISA 7.12.2)
    const int laneN   = lane & 15;
    const int rUpper  = (lane < 16) ? 0 : 8;        // C/D layout M offset

    const uint32_t aBase0 = (uint32_t)(uintptr_t)&Asub[0][0];  // LDS byte offsets
    const uint32_t aBase1 = (uint32_t)(uintptr_t)&Asub[1][0];
    const uint32_t bBase0 = (uint32_t)(uintptr_t)&Bsub[0][0];
    const uint32_t bBase1 = (uint32_t)(uintptr_t)&Bsub[1][0];

    // issue one K-slice's async copies (6 instructions per wave)
    auto issue_slice = [&](int kk, int buf) {
        const uint32_t aB = buf ? aBase1 : aBase0;
        const uint32_t bB = buf ? bBase1 : bBase0;
        // A tile: 64 rows x 32 k f32 = 512 x 16B chunks (2 issues)
        #pragma unroll
        for (int i = 0; i < 2; ++i) {
            const int chunk = i * 256 + tid;
            const int r = chunk >> 3, o = chunk & 7;         // 8 chunks per row
            async_copy16(aB + chunk * 16,
                         &T[(size_t)(rowBase + r) * H_ + kk + (o << 2)]);
        }
        // B tile: 256 cols x 32 k f16 = 1024 x 16B chunks (4 issues)
        #pragma unroll
        for (int i = 0; i < 4; ++i) {
            const int chunk = i * 256 + tid;
            const int c = chunk >> 2, o = chunk & 3;         // 4 chunks per col
            async_copy16(bB + chunk * 16,
                         &Wt[(size_t)c * H_ + kk + (o << 3)]);
        }
    };

    v8f acc[8];
    issue_slice(0, 0);                               // prologue: slice 0 in flight

    #pragma unroll
    for (int t = 0; t < 8; ++t) {                    // load C behind the copies
        const int col = nBase + t * 16 + laneN;
        #pragma unroll
        for (int j = 0; j < 8; ++j)
            acc[t][j] = out[(size_t)(rowBase + mBase + rUpper + j) * H_ + col];
    }

    for (int it = 0; it < 8; ++it) {
        const int kk  = it * 32;
        const int cur = it & 1;

        if (it + 1 < 8) {
            issue_slice(kk + 32, cur ^ 1);           // overlap next slice's DMA
            asm volatile("s_wait_asynccnt 6" ::: "memory");   // cur slice landed
        } else {
            asm volatile("s_wait_asynccnt 0" ::: "memory");
        }
        __syncthreads();                             // cur slice visible to all waves

        const float*    Acur = &Asub[cur][0];
        const _Float16* Bcur = &Bsub[cur][0];

        // A fragment: lane<16 -> row=lane, K {0..7,16..23}; lane>=16 -> row=lane-16, K {8..15,24..31}
        const int arow = mBase + ((lane < 16) ? lane : lane - 16);
        float av[16];
        *(float4*)&av[0]  = *(const float4*)&Acur[arow * 32 + kb0];
        *(float4*)&av[4]  = *(const float4*)&Acur[arow * 32 + kb0 + 4];
        *(float4*)&av[8]  = *(const float4*)&Acur[arow * 32 + kb0 + 16];
        *(float4*)&av[12] = *(const float4*)&Acur[arow * 32 + kb0 + 20];
        v16h afrag;
        #pragma unroll
        for (int i = 0; i < 16; ++i) afrag[i] = (_Float16)av[i];

        v16h bfr[8];
        #pragma unroll
        for (int t = 0; t < 8; ++t) {
            const int col = nBase + t * 16 + laneN;
            v8h blo = *(const v8h*)&Bcur[col * 32 + kb0];
            v8h bhi = *(const v8h*)&Bcur[col * 32 + kb0 + 16];
            bfr[t] = __builtin_shufflevector(blo, bhi, 0,1,2,3,4,5,6,7,8,9,10,11,12,13,14,15);
        }
        #pragma unroll
        for (int t = 0; t < 8; ++t)
            acc[t] = __builtin_amdgcn_wmma_f32_16x16x32_f16(
                false, afrag, false, bfr[t], (short)0, acc[t], false, false);

        __syncthreads();   // cur buffer consumed -> reusable at iteration it+1
    }

    #pragma unroll
    for (int t = 0; t < 8; ++t) {
        const int col = nBase + t * 16 + laneN;
        #pragma unroll
        for (int j = 0; j < 8; ++j)
            out[(size_t)(rowBase + mBase + rUpper + j) * H_ + col] = acc[t][j];
    }
}

// ------------------------------- BatchNorm ---------------------------------

__global__ void __launch_bounds__(256)
bn_stats_kernel(const float* __restrict__ h, float* __restrict__ bsum,
                float* __restrict__ bsq) {
    const int c = threadIdx.x;                       // channel
    const size_t r0 = (size_t)blockIdx.x * 256;      // 256 rows per block
    float s = 0.f, q = 0.f;
    for (int i = 0; i < 256; ++i) {
        float v = h[(r0 + i) * H_ + c];
        s += v; q += v * v;
    }
    atomicAdd(&bsum[c], s);
    atomicAdd(&bsq[c], q);
}

__global__ void bn_finalize_kernel(const float* __restrict__ bsum,
                                   const float* __restrict__ bsq,
                                   const float* __restrict__ g,
                                   const float* __restrict__ be,
                                   float* __restrict__ sc, float* __restrict__ sh) {
    int c = threadIdx.x;                             // <<<1, 256>>>
    const float invN = 1.0f / (float)N_;
    float m   = bsum[c] * invN;
    float var = bsq[c] * invN - m * m;
    float s   = g[c] * rsqrtf(var + 1e-5f);
    sc[c] = s;
    sh[c] = be[c] - m * s;
}

__global__ void __launch_bounds__(256)
bn_apply_lrelu_kernel(float* __restrict__ h, const float* __restrict__ sc,
                      const float* __restrict__ sh) {
    size_t i = (size_t)blockIdx.x * 256 + threadIdx.x;
    int c = (int)(i & (H_ - 1));
    float v = h[i] * sc[c] + sh[c];
    h[i] = v > 0.f ? v : 0.01f * v;
}

// --------------------------- L2 row normalize ------------------------------
__global__ void __launch_bounds__(256) l2norm_kernel(float* __restrict__ h) {
    const int wave = threadIdx.x >> 5;
    const int lane = threadIdx.x & 31;
    const size_t row = (size_t)blockIdx.x * 8 + wave;
    float* p = &h[row * H_ + lane * 8];
    float4 v0 = *(const float4*)(p);
    float4 v1 = *(const float4*)(p + 4);
    float s = v0.x*v0.x + v0.y*v0.y + v0.z*v0.z + v0.w*v0.w
            + v1.x*v1.x + v1.y*v1.y + v1.z*v1.z + v1.w*v1.w;
    #pragma unroll
    for (int off = 16; off > 0; off >>= 1) s += __shfl_xor(s, off, 32);
    float inv = 1.0f / fmaxf(sqrtf(s), 1e-12f);
    v0.x*=inv; v0.y*=inv; v0.z*=inv; v0.w*=inv;
    v1.x*=inv; v1.y*=inv; v1.z*=inv; v1.w*=inv;
    *(float4*)(p)     = v0;
    *(float4*)(p + 4) = v1;
}

// ----------------------------- pooling -------------------------------------
__global__ void __launch_bounds__(256)
pool_kernel(const float* __restrict__ h, const float* __restrict__ x,
            const float* __restrict__ centers, const float* __restrict__ log_temp,
            float* __restrict__ out) {
    __shared__ float a[P_ * NC_];
    __shared__ float colsum[NC_];
    __shared__ float cen[NC_ * 2];
    const int jet  = blockIdx.x;
    const int tid  = threadIdx.x;
    const int base = jet * P_;

    if (tid < NC_ * 2) cen[tid] = centers[tid];
    __syncthreads();

    float lt = fminf(fmaxf(log_temp[0], -2.0f), 3.0f);
    const float Temp = __expf(lt);

    if (tid < P_) {
        float cx = x[(size_t)(base + tid) * IN_ + 0];
        float cy = x[(size_t)(base + tid) * IN_ + 1];
        float s[NC_];
        float m = -1e30f;
        #pragma unroll
        for (int k = 0; k < NC_; ++k) {
            float dx = cx - cen[2 * k], dy = cy - cen[2 * k + 1];
            s[k] = -Temp * (dx * dx + dy * dy);
            m = fmaxf(m, s[k]);
        }
        float sum = 0.f;
        #pragma unroll
        for (int k = 0; k < NC_; ++k) { s[k] = __expf(s[k] - m); sum += s[k]; }
        float inv = 1.0f / sum;
        #pragma unroll
        for (int k = 0; k < NC_; ++k) a[tid * NC_ + k] = s[k] * inv;
    }
    __syncthreads();
    if (tid < NC_) {
        float cs = 0.f;
        for (int n = 0; n < P_; ++n) cs += a[n * NC_ + tid];
        colsum[tid] = cs;
    }
    __syncthreads();

    const int d = tid;                                // 256 threads == H
    float acc[NC_];
    #pragma unroll
    for (int k = 0; k < NC_; ++k) acc[k] = 0.f;
    for (int n = 0; n < P_; ++n) {
        float hv = h[(size_t)(base + n) * H_ + d];    // coalesced over d
        #pragma unroll
        for (int k = 0; k < NC_; ++k) acc[k] += a[n * NC_ + k] * hv;
    }
    #pragma unroll
    for (int k = 0; k < NC_; ++k)
        out[(size_t)jet * NC_ * H_ + (size_t)k * H_ + d] = acc[k] / (colsum[k] + 1e-8f);
}

// ------------------------------- host side ---------------------------------

static void run_cheb_layer256(const float* hin, float* acc, float* s1, float* s2,
                              float* s3, const float* W, const float* bias,
                              _Float16* wh, const int* src, const int* tgt,
                              const float* lw, hipStream_t stream) {
    const int gGemm = N_ / 64;                 // 2400 blocks
    const int gElem = (N_ * H_) / 256;         // 153600 blocks
    const dim3 pgrid(B_, H_ / 32);             // (1024, 8)
    const size_t psh = (size_t)2 * P_ * 32 * sizeof(float);  // 38.4 KB
    const size_t HH = (size_t)H_ * H_;

    wtrans_kernel<<<(6 * (int)HH) / 256, 256, 0, stream>>>(W, wh);  // f32->f16ᵀ
    bias_init_kernel<<<gElem, 256, 0, stream>>>(acc, bias);
    wmma_gemm256<<<gGemm, 256, 0, stream>>>(hin, wh + 0 * HH, acc);      // T0
    prop_kernel<<<pgrid, 256, psh, stream>>>(hin, nullptr, s1, src, tgt, lw, H_, 1.f);
    wmma_gemm256<<<gGemm, 256, 0, stream>>>(s1, wh + 1 * HH, acc);       // T1
    prop_kernel<<<pgrid, 256, psh, stream>>>(s1, hin, s2, src, tgt, lw, H_, 2.f);
    wmma_gemm256<<<gGemm, 256, 0, stream>>>(s2, wh + 2 * HH, acc);       // T2
    prop_kernel<<<pgrid, 256, psh, stream>>>(s2, s1, s3, src, tgt, lw, H_, 2.f);
    wmma_gemm256<<<gGemm, 256, 0, stream>>>(s3, wh + 3 * HH, acc);       // T3
    prop_kernel<<<pgrid, 256, psh, stream>>>(s3, s2, s1, src, tgt, lw, H_, 2.f);
    wmma_gemm256<<<gGemm, 256, 0, stream>>>(s1, wh + 4 * HH, acc);       // T4
    prop_kernel<<<pgrid, 256, psh, stream>>>(s1, s3, s2, src, tgt, lw, H_, 2.f);
    wmma_gemm256<<<gGemm, 256, 0, stream>>>(s2, wh + 5 * HH, acc);       // T5
}

static void run_bn_lrelu(float* h, const float* g, const float* be, float* bsum,
                         float* bsq, float* bsc, float* bsh, hipStream_t stream) {
    zero_kernel<<<(2 * H_ + 255) / 256, 256, 0, stream>>>(bsum, 2 * H_);  // bsq contiguous
    bn_stats_kernel<<<N_ / 256, 256, 0, stream>>>(h, bsum, bsq);
    bn_finalize_kernel<<<1, H_, 0, stream>>>(bsum, bsq, g, be, bsc, bsh);
    bn_apply_lrelu_kernel<<<(N_ * H_) / 256, 256, 0, stream>>>(h, bsc, bsh);
}

extern "C" void kernel_launch(void* const* d_in, const int* in_sizes, int n_in,
                              void* d_out, int out_size, void* d_ws, size_t ws_size,
                              hipStream_t stream) {
    (void)in_sizes; (void)n_in; (void)out_size; (void)ws_size;

    const float* x   = (const float*)d_in[0];
    const int*   ei  = (const int*)d_in[1];
    const int*   src = ei;
    const int*   tgt = ei + E_;
    const float* W1  = (const float*)d_in[3];
    const float* b1  = (const float*)d_in[4];
    const float* W2  = (const float*)d_in[5];
    const float* b2  = (const float*)d_in[6];
    const float* W3  = (const float*)d_in[7];
    const float* b3  = (const float*)d_in[8];
    const float* W4  = (const float*)d_in[9];
    const float* b4  = (const float*)d_in[10];
    const float* g1  = (const float*)d_in[11];
    const float* be1 = (const float*)d_in[12];
    const float* g2  = (const float*)d_in[13];
    const float* be2 = (const float*)d_in[14];
    const float* g3  = (const float*)d_in[15];
    const float* be3 = (const float*)d_in[16];
    const float* cen = (const float*)d_in[17];
    const float* ltp = (const float*)d_in[18];
    float* out = (float*)d_out;

    // workspace (~641 MB): 4 x [N,H] rotating buffers + lw + stats + f16 W
    const size_t NH = (size_t)N_ * H_;
    float* buf0 = (float*)d_ws;
    float* buf1 = buf0 + NH;
    float* buf2 = buf1 + NH;
    float* buf3 = buf2 + NH;
    float* lw   = buf3 + NH;
    float* deg  = lw + E_;
    float* bsum = deg + N_;
    float* bsq  = bsum + H_;
    float* bsc  = bsq + H_;
    float* bsh  = bsc + H_;
    _Float16* wh = (_Float16*)(bsh + H_);   // 6*H*H halfs (786 KB)

    // ---- scaled Laplacian edge weights ----
    zero_kernel<<<(N_ + 255) / 256, 256, 0, stream>>>(deg, N_);
    deg_kernel<<<(E_ + 255) / 256, 256, 0, stream>>>(src, tgt, deg);
    lapw_kernel<<<(E_ + 255) / 256, 256, 0, stream>>>(src, tgt, deg, lw);

    // ---- layer 1 (IN=3 -> H), scalar GEMM path ----
    {
        const int gElem = (N_ * H_) / 256;
        const dim3 pg3(B_, 1);
        const size_t sh3 = (size_t)2 * P_ * IN_ * sizeof(float);
        bias_init_kernel<<<gElem, 256, 0, stream>>>(buf0, b1);
        gemm3_kernel<<<gElem, 256, 0, stream>>>(x, W1 + 0 * IN_ * H_, buf0);          // T0=x
        prop_kernel<<<pg3, 256, sh3, stream>>>(x, nullptr, buf1, src, tgt, lw, IN_, 1.f);
        gemm3_kernel<<<gElem, 256, 0, stream>>>(buf1, W1 + 1 * IN_ * H_, buf0);       // T1
        prop_kernel<<<pg3, 256, sh3, stream>>>(buf1, x, buf2, src, tgt, lw, IN_, 2.f);
        gemm3_kernel<<<gElem, 256, 0, stream>>>(buf2, W1 + 2 * IN_ * H_, buf0);       // T2
        prop_kernel<<<pg3, 256, sh3, stream>>>(buf2, buf1, buf3, src, tgt, lw, IN_, 2.f);
        gemm3_kernel<<<gElem, 256, 0, stream>>>(buf3, W1 + 3 * IN_ * H_, buf0);       // T3
        prop_kernel<<<pg3, 256, sh3, stream>>>(buf3, buf2, buf1, src, tgt, lw, IN_, 2.f);
        gemm3_kernel<<<gElem, 256, 0, stream>>>(buf1, W1 + 4 * IN_ * H_, buf0);       // T4
        prop_kernel<<<pg3, 256, sh3, stream>>>(buf1, buf3, buf2, src, tgt, lw, IN_, 2.f);
        gemm3_kernel<<<gElem, 256, 0, stream>>>(buf2, W1 + 5 * IN_ * H_, buf0);       // T5
    }
    run_bn_lrelu(buf0, g1, be1, bsum, bsq, bsc, bsh, stream);

    // ---- layers 2..4 (H -> H), WMMA path ----
    run_cheb_layer256(buf0, buf1, buf2, buf3, buf0, W2, b2, wh, src, tgt, lw, stream);
    run_bn_lrelu(buf1, g2, be2, bsum, bsq, bsc, bsh, stream);

    run_cheb_layer256(buf1, buf0, buf2, buf3, buf1, W3, b3, wh, src, tgt, lw, stream);
    run_bn_lrelu(buf0, g3, be3, bsum, bsq, bsc, bsh, stream);

    run_cheb_layer256(buf0, buf1, buf2, buf3, buf0, W4, b4, wh, src, tgt, lw, stream);

    // ---- L2 normalize rows of final h (buf1) ----
    l2norm_kernel<<<N_ / 8, 256, 0, stream>>>(buf1);

    // ---- distance-softmax pooling -> d_out [B, NC*H] ----
    pool_kernel<<<B_, 256, 0, stream>>>(buf1, x, cen, ltp, out);
}
